// DropGCN_27590869909517
// MI455X (gfx1250) — matrix operand order repaired and checked by reading, not compiled, not run on hardware
//
#include <hip/hip_runtime.h>
#include <stdint.h>

// ---------------------------------------------------------------------------
// DropGCN on MI455X (gfx1250, wave32).
//   out = relu( mean_r( diag(keep_r) A diag(keep_r) X ) W^T + b )
// HBM-bound on one streaming pass over A (256 MB ~= 11 us @ 23.3 TB/s).
//  * bf16 WMMA 16x16x32, f32 accumulate: 10 wmma per wave per K=32 chunk.
//  * Double-buffered LDS; staging by the Tensor Data Mover
//    (tensor_load_to_lds + s_wait_tensorcnt), using TDM's pad feature to
//    produce bank-conflict-free LDS pitches in hardware. Fallback to plain
//    vector copies if the TDM builtin is unavailable.
//  * K split 8 ways for occupancy; deterministic fixed-order reduction.
// ---------------------------------------------------------------------------

typedef __bf16 bf16;
typedef bf16  v8bf  __attribute__((ext_vector_type(8)));
typedef bf16  v16bf __attribute__((ext_vector_type(16)));
typedef float v8f   __attribute__((ext_vector_type(8)));
typedef unsigned int u32x4 __attribute__((ext_vector_type(4)));
typedef int          i32x4 __attribute__((ext_vector_type(4)));
typedef int          i32x8 __attribute__((ext_vector_type(8)));

#ifndef __has_builtin
#define __has_builtin(x) 0
#endif
#if __has_builtin(__builtin_amdgcn_tensor_load_to_lds) && \
    __has_builtin(__builtin_amdgcn_s_wait_tensorcnt)
#define USE_TDM 1
#else
#define USE_TDM 0
#endif

#define NN    8192
#define DIN   32
#define NR    5
#define BM    128        // rows per block (8 waves x 16-row M-tile)
#define BK    32         // K chunk == WMMA K
#define KS    8          // K slices (grid.y)
#define KSEG  (NN / KS)  // 1024 columns per slice
#define APITCH 36        // A LDS pitch in f32  (144 B: 32 data + 4 pad dwords)
#define XPITCH 40        // X LDS pitch in bf16 (80 B: 64 B data + 16 B pad)

// ---- Kernel 1: XmT[chunk][r][n][k] = keep[r][j] ? bf16(X[j][n]) : 0 -------
// (j = chunk*32+k). Chunk-major so each K-chunk's 10 KB slab is contiguous
// (one TDM 2-D descriptor per chunk); rows K-major for the WMMA B layout.
__global__ __launch_bounds__(256) void prep_xm(const float* __restrict__ X,
                                               const uint8_t* __restrict__ keep,
                                               bf16* __restrict__ XmT) {
    const int idx = blockIdx.x * 256 + threadIdx.x;   // < 5*8192*32 = 5<<18
    const int k   = idx & 31;
    const int n   = (idx >> 5) & 31;
    const int sub = idx >> 10;          // 0 .. 5*256-1 == chunk*5 + r
    const int r   = sub % NR;
    const int chunk = sub / NR;
    if (sub < NR * (NN / BK)) {
        const int j = chunk * 32 + k;
        const float v = keep[r * NN + j] ? X[j * DIN + n] : 0.0f;
        XmT[idx] = (bf16)v;
    }
}

// ---- TDM 2-D tile load: global -> LDS with hardware padding ---------------
// D# per CDNA5 ISA. data_size = 4 bytes.
__device__ __forceinline__ void tdm_load_2d(uint32_t ldsOff, const void* gsrc,
                                            uint32_t tileW, uint32_t tileH,
                                            uint32_t strideElems,
                                            uint32_t padIntCode,   // 2^code * 8 B
                                            uint32_t padAmtCode) { // (code+1) dwords
#if USE_TDM
    const uint64_t ga = (uint64_t)(uintptr_t)gsrc;
    // group0: count=1 (valid user D#), lds_addr, global_addr[56:0], type=2
    u32x4 g0 = { 1u, ldsOff, (uint32_t)ga,
                 (uint32_t)(ga >> 32) | (2u << 30) };
    // group1
    const uint32_t w0  = (2u << 16)                    // data_size = 4 B
                       | (1u << 20)                    // pad_enable
                       | (padIntCode << 22) | (padAmtCode << 25);
    const uint32_t td0 = strideElems;                  // tensor_dim0 >= tile W
    const uint32_t td1 = 1u << 20;                     // tensor_dim1 (large)
    i32x8 g1 = { (int)w0,
                 (int)((td0 & 0xffffu) << 16),                          // W1
                 (int)(((td0 >> 16) & 0xffffu) | ((td1 & 0xffffu) << 16)),
                 (int)(((td1 >> 16) & 0xffffu) | (tileW << 16)),        // W3
                 (int)(tileH & 0xffffu),               // tile_dim1, tile_dim2=0
                 (int)strideElems,                     // tensor_dim0_stride lo32
                 0, 0 };
    i32x4 g2 = { 1, 1, 0, 0 };                         // tensor_dim2/3 = 1
    i32x4 g3 = { 0, 0, 0, 0 };
    i32x8 g4 = { 0, 0, 0, 0, 0, 0, 0, 0 };             // extra group (clang-23
                                                       // 6-arg form), zero-fill
    __builtin_amdgcn_tensor_load_to_lds(g0, g1, g2, g3, g4, 0);
#endif
}

// ---- Kernel 2: P[ks][i][d] = 0.2 * sum_r keep[r][i] * (A_slice @ Xm_r) ----
__global__ __launch_bounds__(256) void gcn_main(const float* __restrict__ A,
                                                const bf16* __restrict__ XmT,
                                                const uint8_t* __restrict__ keep,
                                                float* __restrict__ P) {
    __shared__ __align__(16) float Asf[2][BM][APITCH];        // 36.9 KB, f32
    __shared__ __align__(16) bf16  Xs[2][NR][DIN][XPITCH];    // 25.6 KB, bf16

    const int t       = threadIdx.x;
    const int lane    = t & 31;
    const int w       = t >> 5;            // wave id 0..7 -> 16-row M-tile
    const int l15     = lane & 15;
    const int lhalf   = (lane < 16) ? 0 : 1;
    const int rowBase = blockIdx.x * BM;
    const int ksl     = blockIdx.y;
    const int koBeg   = ksl * KSEG;
    const int koEnd   = koBeg + KSEG;

    v8f acc[NR][2];
    const v8f vzero = {0.f, 0.f, 0.f, 0.f, 0.f, 0.f, 0.f, 0.f};
#pragma unroll
    for (int r = 0; r < NR; ++r) { acc[r][0] = vzero; acc[r][1] = vzero; }

    // ---- staging: two TDM descriptors per chunk (wave 0), or plain copies -
    auto stage = [&](int b, int ko) {
#if USE_TDM
        if (w == 0) {
            // A tile: 128 rows x 32 f32, row stride NN; LDS pad 128 B + 16 B
            tdm_load_2d((uint32_t)(uintptr_t)&Asf[b][0][0],
                        &A[(size_t)rowBase * NN + ko],
                        /*tileW=*/BK, /*tileH=*/BM, /*stride=*/NN,
                        /*padInt 128B=*/4, /*padAmt 16B=*/3);
            // X slab: 160 rows x 64 B (16 x 4B), contiguous; pad 64 B + 16 B
            tdm_load_2d((uint32_t)(uintptr_t)&Xs[b][0][0][0],
                        &XmT[(size_t)(ko >> 5) * (NR * DIN * BK)],
                        /*tileW=*/16, /*tileH=*/NR * DIN, /*stride=*/16,
                        /*padInt 64B=*/3, /*padAmt 16B=*/3);
        }
#else
        {   // A tile: pure f32 copy (64 B per thread)
            const int srow = t >> 1, shalf = t & 1;
            const size_t ga = (size_t)(rowBase + srow) * NN + ko + shalf * 16;
            float4 f0 = *reinterpret_cast<const float4*>(&A[ga + 0]);
            float4 f1 = *reinterpret_cast<const float4*>(&A[ga + 4]);
            float4 f2 = *reinterpret_cast<const float4*>(&A[ga + 8]);
            float4 f3 = *reinterpret_cast<const float4*>(&A[ga + 12]);
            if (ko + BK < koEnd)
                __builtin_prefetch(&A[ga + BK], 0, 3);   // global_prefetch_b8
            float* dst = &Asf[b][srow][shalf * 16];
            *reinterpret_cast<float4*>(dst + 0)  = f0;
            *reinterpret_cast<float4*>(dst + 4)  = f1;
            *reinterpret_cast<float4*>(dst + 8)  = f2;
            *reinterpret_cast<float4*>(dst + 12) = f3;
        }
        if (t < NR * DIN) {   // X slab: bf16 copy, one (r,n) row per thread
            const int xr = t >> 5, xn = t & 31;
            const bf16* src =
                &XmT[(size_t)((((ko >> 5) * NR + xr) * DIN + xn)) * BK];
            bf16* dst = &Xs[b][xr][xn][0];
            *reinterpret_cast<v8bf*>(dst + 0)  = *reinterpret_cast<const v8bf*>(src + 0);
            *reinterpret_cast<v8bf*>(dst + 8)  = *reinterpret_cast<const v8bf*>(src + 8);
            *reinterpret_cast<v8bf*>(dst + 16) = *reinterpret_cast<const v8bf*>(src + 16);
            *reinterpret_cast<v8bf*>(dst + 24) = *reinterpret_cast<const v8bf*>(src + 24);
        }
#endif
    };

    stage(0, koBeg);
    int buf = 0;
    for (int ko = koBeg; ko < koEnd; ko += BK) {
#if USE_TDM
        if (w == 0) __builtin_amdgcn_s_wait_tensorcnt(0);
#endif
        __syncthreads();                       // staged buffer now visible
        if (ko + BK < koEnd) stage(buf ^ 1, ko + BK);   // overlap next chunk

        // --- A fragment: f32 LDS -> bf16 regs (v_cvt_pk_bf16_f32) ----------
        // 16-bit A 16x32 layout: lane M=l15; lanes<16: K 0..7|16..23,
        //                        lanes>=16: K 8..15|24..31.
        const float* arowp = &Asf[buf][w * 16 + l15][0];
        const int kbA = lhalf * 8;
        const float4 q0 = *reinterpret_cast<const float4*>(arowp + kbA);
        const float4 q1 = *reinterpret_cast<const float4*>(arowp + kbA + 4);
        const float4 q2 = *reinterpret_cast<const float4*>(arowp + 16 + kbA);
        const float4 q3 = *reinterpret_cast<const float4*>(arowp + 16 + kbA + 4);
        const v16bf a = {(bf16)q0.x, (bf16)q0.y, (bf16)q0.z, (bf16)q0.w,
                         (bf16)q1.x, (bf16)q1.y, (bf16)q1.z, (bf16)q1.w,
                         (bf16)q2.x, (bf16)q2.y, (bf16)q2.z, (bf16)q2.w,
                         (bf16)q3.x, (bf16)q3.y, (bf16)q3.z, (bf16)q3.w};

        // --- 10 x v_wmma_f32_16x16x32_bf16 ---------------------------------
        // 16-bit B 32x16 layout: lane N=l15; lanes<16: K 0..7|16..23, >=16: K 8..15|24..31.
        const int kbB = lhalf * 8;
#pragma unroll
        for (int r = 0; r < NR; ++r) {
#pragma unroll
            for (int h = 0; h < 2; ++h) {
                const bf16* bp = &Xs[buf][r][h * 16 + l15][kbB];
                const v8bf blo = *reinterpret_cast<const v8bf*>(bp);
                const v8bf bhi = *reinterpret_cast<const v8bf*>(bp + 16);
                const v16bf bb = __builtin_shufflevector(blo, bhi,
                    0, 1, 2, 3, 4, 5, 6, 7, 8, 9, 10, 11, 12, 13, 14, 15);
                acc[r][h] = __builtin_amdgcn_wmma_f32_16x16x32_bf16(
                    false, a, false, bb, (short)0, acc[r][h], false, false);
            }
        }
        buf ^= 1;
    }

    // --- epilogue: output-row keep mask + mean over replicas ---------------
#pragma unroll
    for (int g = 0; g < 8; ++g) {
        const int rloc = lhalf * 8 + g;
        const int grow = rowBase + w * 16 + rloc;
        float s0 = 0.f, s1 = 0.f;
#pragma unroll
        for (int r = 0; r < NR; ++r) {
            const float km = keep[(size_t)r * NN + grow] ? 0.2f : 0.0f;
            s0 += km * acc[r][0][g];
            s1 += km * acc[r][1][g];
        }
        float* prow = &P[((size_t)ksl * NN + grow) * DIN];
        prow[l15]      = s0;
        prow[16 + l15] = s1;
    }
}

// ---- Kernel 3: deterministic K-slice reduction + FC + ReLU ----------------
__global__ __launch_bounds__(256) void fc_relu(const float* __restrict__ P,
                                               const float* __restrict__ W,
                                               const float* __restrict__ bias,
                                               float* __restrict__ out) {
    __shared__ float Ms[8][DIN];
    __shared__ float Ws[DIN * DIN];

    const int t = threadIdx.x;
    const int rowBase = blockIdx.x * 8;
#pragma unroll
    for (int i = 0; i < 4; ++i)
        Ws[t + i * 256] = W[t + i * 256];

    const int lr = t >> 5;
    const int d  = t & 31;
    float s = 0.f;
#pragma unroll
    for (int ks = 0; ks < KS; ++ks)   // fixed-order sum -> deterministic
        s += P[((size_t)ks * NN + rowBase + lr) * DIN + d];
    Ms[lr][d] = s;
    __syncthreads();

    const int dout = t & 31;
    float a = bias[dout];
#pragma unroll
    for (int din = 0; din < DIN; ++din)
        a += Ms[lr][din] * Ws[dout * DIN + din];
    out[(size_t)(rowBase + lr) * DIN + dout] = fmaxf(a, 0.f);
}

// ---------------------------------------------------------------------------
extern "C" void kernel_launch(void* const* d_in, const int* in_sizes, int n_in,
                              void* d_out, int out_size, void* d_ws, size_t ws_size,
                              hipStream_t stream) {
    const float*   A    = (const float*)d_in[0];
    const float*   X    = (const float*)d_in[1];
    const float*   W    = (const float*)d_in[2];
    const float*   b    = (const float*)d_in[3];
    const uint8_t* keep = (const uint8_t*)d_in[4];
    float*         out  = (float*)d_out;

    bf16*  XmT = (bf16*)d_ws;                         // 5*8192*32 bf16 = 2.62 MB
    float* P   = (float*)((char*)d_ws + (size_t)NR * NN * DIN * sizeof(bf16));

    prep_xm<<<(NR * NN * DIN) / 256, 256, 0, stream>>>(X, keep, XmT);
    dim3 g2(NN / BM, KS);
    gcn_main<<<g2, 256, 0, stream>>>(A, XmT, keep, P);
    fc_relu<<<NN / 8, 256, 0, stream>>>(P, W, b, out);
}
